// DkmCompGraph_64080912056405
// MI455X (gfx1250) — matrix operand order repaired and checked by reading, not compiled, not run on hardware
//
#include <hip/hip_runtime.h>
#include <hip/hip_bf16.h>
#include <stdint.h>

#define N_PTS 32768
#define DIM   512
#define KCL   1024

typedef __attribute__((ext_vector_type(16))) __bf16 v16bf;
typedef __attribute__((ext_vector_type(8)))  float  v8f;

// ---------------------------------------------------------------------------
// Kernel 1: convert cluster_rep (K x D, f32) to split-bf16 hi/lo and compute
// per-cluster squared norms. One block per cluster row.
// ---------------------------------------------------------------------------
__global__ void dkm_prep(const float* __restrict__ crep,
                         __bf16* __restrict__ chi,
                         __bf16* __restrict__ clo,
                         float*  __restrict__ cnorm) {
  __shared__ float red[128];
  const int k = blockIdx.x;
  const int t = threadIdx.x;
  const size_t o = (size_t)k * DIM + t * 4;
  float4 v = *(const float4*)(crep + o);
  float e[4] = {v.x, v.y, v.z, v.w};
  float ss = 0.0f;
#pragma unroll
  for (int j = 0; j < 4; ++j) {
    ss += e[j] * e[j];
    __bf16 h = (__bf16)e[j];
    chi[o + j] = h;
    clo[o + j] = (__bf16)(e[j] - (float)h);
  }
  red[t] = ss;
  __syncthreads();
#pragma unroll
  for (int s = 64; s > 0; s >>= 1) {
    if (t < s) red[t] += red[t + s];
    __syncthreads();
  }
  if (t == 0) cnorm[k] = red[0];
}

// ---------------------------------------------------------------------------
// Kernel 2: fused distances + online softmin + argmin.
// Block = 256 threads (8 wave32s). Each wave owns a 16-row tile of points,
// keeps its split-bf16 A operands resident in VGPRs, and sweeps all 1024
// clusters in 16-column WMMA tiles. Three INDEPENDENT accumulators (hi*hi,
// hi*lo, lo*hi) break the WMMA RAW chain so the XDL pipe can stay full.
// C/D layout: lane n (n<16) = column n, VGPR r = row r (lanes 16-31: row r+8).
// ---------------------------------------------------------------------------
__global__ __launch_bounds__(256, 1) void dkm_main(
    const float*  __restrict__ inp,
    const __bf16* __restrict__ chi,
    const __bf16* __restrict__ clo,
    const float*  __restrict__ cnorm,
    float*        __restrict__ out,      // [N*D centroids][N idx][1 loss]
    float*        __restrict__ rowloss)  // [N] per-row loss (pre 1/K)
{
  __shared__ float4 st[8][16][16];
  __shared__ float  xns[8][32];

  const int tid   = threadIdx.x;
  const int wave  = tid >> 5;
  const int lane  = tid & 31;
  const int laneR = lane & 15;   // A: point row loaded; B/C: column within tile
  const int half  = lane >> 4;   // 0 = lanes 0-15, 1 = lanes 16-31
  const int rowbase = blockIdx.x * 128 + wave * 16;

  // ---- load + split-convert A tile (16 rows x 512) into registers ----
  // A-layout (16-bit, 16x32): lanes 0-15 hold K=0..15 of row=lane,
  // lanes 16-31 hold K=16..31 of row=lane-16 (matches B's contiguous order).
  v16bf Ahi[16], Alo[16];
  const float* arow = inp + (size_t)(rowbase + laneR) * DIM;
  const int koff = half ? 16 : 0;
  float xn = 0.0f;
#pragma unroll
  for (int s = 0; s < 16; ++s) {
    const float* p0 = arow + s * 32 + koff;
    const float* p1 = arow + s * 32 + 8 + koff;
    float4 a0 = *(const float4*)(p0);
    float4 a1 = *(const float4*)(p0 + 4);
    float4 a2 = *(const float4*)(p1);
    float4 a3 = *(const float4*)(p1 + 4);
    float e[16] = {a0.x, a0.y, a0.z, a0.w, a1.x, a1.y, a1.z, a1.w,
                   a2.x, a2.y, a2.z, a2.w, a3.x, a3.y, a3.z, a3.w};
#pragma unroll
    for (int j = 0; j < 16; ++j) {
      xn += e[j] * e[j];
      __bf16 h = (__bf16)e[j];
      Ahi[s][j] = h;
      Alo[s][j] = (__bf16)(e[j] - (float)h);
    }
  }
  xns[wave][lane] = xn;  // row laneR partial norm (two halves per row)

  // ---- per-lane online softmin state: 8 rows x (this lane's columns) ----
  float mn[8], lsum[8], ssum[8];
  int   bidx[8];
#pragma unroll
  for (int r = 0; r < 8; ++r) {
    mn[r] = 3.4e38f; lsum[r] = 0.0f; ssum[r] = 0.0f; bidx[r] = 0;
  }

  // ---- sweep all clusters in 16-wide tiles ----
  for (int kt = 0; kt < KCL / 16; ++kt) {
    const int cbase = kt * 16;
    const int mycol = cbase + laneR;
    const float cn = cnorm[mycol];
    // B-layout used here, same k order as A (16-bit, 32x16): lane n holds column n, K=0..15;
    // lane n+16 holds column n, K=16..31 -> contiguous 16 bf16 per lane.
    const __bf16* bh = chi + (size_t)mycol * DIM + half * 16;
    const __bf16* bl = clo + (size_t)mycol * DIM + half * 16;
    v8f acc0 = {};  // hi * hi
    v8f acc1 = {};  // hi * lo
    v8f acc2 = {};  // lo * hi
#pragma unroll
    for (int s = 0; s < 16; ++s) {
      v16bf Bh = *(const v16bf*)(bh + s * 32);
      v16bf Bl = *(const v16bf*)(bl + s * 32);
      acc0 = __builtin_amdgcn_wmma_f32_16x16x32_bf16(false, Ahi[s], false, Bh,
                                                     (short)0, acc0, false, false);
      acc1 = __builtin_amdgcn_wmma_f32_16x16x32_bf16(false, Ahi[s], false, Bl,
                                                     (short)0, acc1, false, false);
      acc2 = __builtin_amdgcn_wmma_f32_16x16x32_bf16(false, Alo[s], false, Bh,
                                                     (short)0, acc2, false, false);
    }
    // y = ||c||^2 - 2 x.c  (row-constant ||x||^2 dropped: softmin-invariant)
#pragma unroll
    for (int r = 0; r < 8; ++r) {
      float g  = acc0[r] + (acc1[r] + acc2[r]);
      float y  = cn - 2.0f * g;
      float nm = fminf(mn[r], y);
      float fo = __expf(nm - mn[r]);   // rescale old accumulators
      float en = __expf(nm - y);       // new term
      lsum[r] = lsum[r] * fo + en;
      ssum[r] = ssum[r] * fo + en * y;
      if (y < mn[r]) bidx[r] = mycol;  // columns increase -> first-min kept
      mn[r] = nm;
    }
  }

  // ---- merge the 16 column-lanes of each row ----
#pragma unroll
  for (int r = 0; r < 8; ++r) {
    st[wave][r + half * 8][laneR] =
        make_float4(mn[r], lsum[r], ssum[r], (float)bidx[r]);
  }
  __syncthreads();

  if (lane < 16) {
    const int row = lane;
    float bm = 3.4e38f, bl_ = 0.0f, bs = 0.0f, bif = 1.0e9f;
#pragma unroll
    for (int c = 0; c < 16; ++c) {
      float4 t = st[wave][row][c];
      float nm = fminf(bm, t.x);
      float f1 = __expf(nm - bm);
      float f2 = __expf(nm - t.x);
      bl_ = bl_ * f1 + t.y * f2;
      bs  = bs  * f1 + t.z * f2;
      if (t.x < bm || (t.x == bm && t.w < bif)) bif = t.w;  // first-index tie
      bm = nm;
    }
    const float xr = xns[wave][row] + xns[wave][row + 16];
    const int grow = rowbase + row;
    out[(size_t)N_PTS * DIM + grow] = bif;      // idx output (as float)
    rowloss[grow] = xr + bs / bl_;              // sum_k soft*d2 for this row
  }
}

// ---------------------------------------------------------------------------
// Kernel 3: deterministic fixed-order reduction of per-row losses.
// ---------------------------------------------------------------------------
__global__ void dkm_loss_reduce(const float* __restrict__ rowloss,
                                float* __restrict__ loss_slot) {
  __shared__ float red[256];
  const int t = threadIdx.x;
  float s = 0.0f;
  for (int i = t; i < N_PTS; i += 256) s += rowloss[i];
  red[t] = s;
  __syncthreads();
#pragma unroll
  for (int k = 128; k > 0; k >>= 1) {
    if (t < k) red[t] += red[t + k];
    __syncthreads();
  }
  if (t == 0) *loss_slot = red[0] * (1.0f / KCL);
}

// ---------------------------------------------------------------------------
// Kernel 4: centroids gather: out[row] = cluster_rep[idx[row]].
// One block per point; cluster_rep (2 MB) is L2-resident.
// ---------------------------------------------------------------------------
__global__ void dkm_gather(const float* __restrict__ crep,
                           float* __restrict__ out) {
  const int row = blockIdx.x;
  const int t = threadIdx.x;
  const int k = (int)out[(size_t)N_PTS * DIM + row];
  float4 v = *(const float4*)(crep + (size_t)k * DIM + t * 4);
  *(float4*)(out + (size_t)row * DIM + t * 4) = v;
}

// ---------------------------------------------------------------------------
extern "C" void kernel_launch(void* const* d_in, const int* in_sizes, int n_in,
                              void* d_out, int out_size, void* d_ws,
                              size_t ws_size, hipStream_t stream) {
  (void)in_sizes; (void)n_in; (void)out_size; (void)ws_size;
  const float* inp  = (const float*)d_in[0];   // (N, D) f32
  const float* crep = (const float*)d_in[1];   // (K, D) f32
  float* out = (float*)d_out;

  // workspace layout: chi 1MB | clo 1MB | cnorm 4KB | rowloss 128KB
  char* ws = (char*)d_ws;
  __bf16* chi    = (__bf16*)(ws);
  __bf16* clo    = (__bf16*)(ws + (size_t)KCL * DIM * 2);
  float*  cnorm  = (float*)(ws + (size_t)KCL * DIM * 4);
  float*  rowls  = (float*)(ws + (size_t)KCL * DIM * 4 + KCL * 4);
  float*  loss   = out + (size_t)N_PTS * DIM + N_PTS;

  dkm_prep<<<KCL, 128, 0, stream>>>(crep, chi, clo, cnorm);
  dkm_main<<<N_PTS / 128, 256, 0, stream>>>(inp, chi, clo, cnorm, out, rowls);
  dkm_loss_reduce<<<1, 256, 0, stream>>>(rowls, loss);
  dkm_gather<<<N_PTS, 128, 0, stream>>>(crep, out);
}
